// SS2D_4423816315297
// MI455X (gfx1250) — matrix operand order repaired and checked by reading, not compiled, not run on hardware
//
#include <hip/hip_runtime.h>
#include <hip/hip_bf16.h>
#include <cstddef>

// ---------- problem constants ----------
#define BB 2
#define HH 64
#define WW 64
#define LL (HH*WW)          // 4096
#define DM 96               // d_model
#define DI 192              // d_inner
#define KK 4
#define NS 16               // d_state
#define RK 6                // dt_rank
#define CPAD 48             // 38 padded to 3x16 tiles
#define MROWS (BB*LL)       // 8192

// ---------- workspace layout (bytes) ----------
#define OFF_XBF   ((size_t)0)                       // bf16 x        [8192,96]
#define OFF_WIN   (OFF_XBF  + (size_t)MROWS*DM*2)   // bf16 in_proj_w[384,96]
#define OFF_XPW   (OFF_WIN  + (size_t)2*DI*DM*2)    // bf16 x_proj_w padded [4,48,192]
#define OFF_WOUT  (OFF_XPW  + (size_t)KK*CPAD*DI*2) // bf16 out_proj_w[96,192]
#define OFF_XZ    (OFF_WOUT + (size_t)DM*DI*2)      // f32 xz [8192,384]
#define OFF_XS    (OFF_XZ   + (size_t)MROWS*2*DI*4) // bf16 xs [2,4,4096,192]
#define OFF_XDBL  (OFF_XS   + (size_t)BB*KK*LL*DI*2)// f32 xdbl [2,4,4096,48]
#define OFF_OUTY  (OFF_XDBL + (size_t)BB*KK*LL*CPAD*4) // f32 out_y [2,4,4096,192]
#define OFF_YLN   (OFF_OUTY + (size_t)BB*KK*LL*DI*4)   // bf16 yln [8192,192]

#define AS1 __attribute__((address_space(1)))
#define AS3 __attribute__((address_space(3)))

typedef int v4i __attribute__((ext_vector_type(4)));

#if __has_builtin(__builtin_amdgcn_global_load_async_to_lds_b128)
#define HAVE_ASYNC_LDS 1
#else
#define HAVE_ASYNC_LDS 0
#endif

template<int N>
__device__ __forceinline__ void wait_async() {
#if HAVE_ASYNC_LDS
#if __has_builtin(__builtin_amdgcn_s_wait_asynccnt)
    __builtin_amdgcn_s_wait_asynccnt(N);
#else
    asm volatile("s_wait_asynccnt %0" :: "i"(N) : "memory");
#endif
#endif
}

#if HAVE_ASYNC_LDS
// async global->LDS copy of one 16-byte chunk (per enabled lane)
__device__ __forceinline__ void async_copy16(const float* gsrc, float* lds_dst) {
    __builtin_amdgcn_global_load_async_to_lds_b128(
        (AS1 v4i*)(v4i*)const_cast<float*>(gsrc),
        (AS3 v4i*)(v4i*)lds_dst, 0, 0);
}
#endif

// ---------- WMMA fragment helpers (wave32, 16x16x32 bf16) ----------
typedef __attribute__((ext_vector_type(16))) __bf16 v16bf;
typedef __attribute__((ext_vector_type(8)))  float  v8f;

union FragBF { v16bf v; unsigned u[8]; };

// A: row-major [16, K] bf16 tile base (lda even). B: row-major [16, K] (i.e. W[N,K]).
// Per CDNA5 ISA 7.12.2 layouts:
//  A 16x32: lane m=lane&15, khalf=lane>>4; vgpr i holds K pair at (i>>2)*16 + khalf*8 + (i&3)*2
//  B 32x16: lane n=lane&15, khalf=lane>>4; vgpr i holds K pair at (i>>2)*16 + khalf*8 + (i&3)*2
template<int KSTEPS>
__device__ __forceinline__ v8f wmma_tile_bf16(const __bf16* __restrict__ A, int lda,
                                              const __bf16* __restrict__ Bt, int ldb) {
    const int lane = threadIdx.x & 31;
    const int half = lane >> 4;
    const int mn   = lane & 15;
    v8f acc = {};
#pragma unroll
    for (int ks = 0; ks < KSTEPS; ++ks) {
        const int k0 = ks * 32;
        FragBF a, b;
#pragma unroll
        for (int i = 0; i < 8; ++i) {
            const int ka = k0 + ((i >> 2) << 4) + (half << 3) + ((i & 3) << 1);
            a.u[i] = *(const unsigned*)(A + (size_t)mn * lda + ka);
            const int kb = k0 + ((i >> 2) << 4) + (half << 3) + ((i & 3) << 1);
            b.u[i] = *(const unsigned*)(Bt + (size_t)mn * ldb + kb);
        }
        acc = __builtin_amdgcn_wmma_f32_16x16x32_bf16(false, a.v, false, b.v,
                                                      (short)0, acc, false, false);
    }
    return acc;
}

// D 16x16 f32: vgpr r -> row m = r + 8*(lane>>4), col n = lane&15
__device__ __forceinline__ void store_tile_f32(float* __restrict__ C, int ldc, v8f acc) {
    const int lane = threadIdx.x & 31;
    const int half = lane >> 4;
    const int n    = lane & 15;
#pragma unroll
    for (int r = 0; r < 8; ++r) C[(size_t)(r + half * 8) * ldc + n] = acc[r];
}

// ---------- kernel 1: cast/pad weights + activations to bf16 ----------
__global__ void prep_cast(const float* __restrict__ x, const float* __restrict__ win,
                          const float* __restrict__ xpw, const float* __restrict__ wout,
                          __bf16* __restrict__ xbf, __bf16* __restrict__ winbf,
                          __bf16* __restrict__ xpwbf, __bf16* __restrict__ woutbf) {
    const int i = blockIdx.x * blockDim.x + threadIdx.x;
    if (i < MROWS * DM) xbf[i] = (__bf16)x[i];
    if (i < 2 * DI * DM) winbf[i] = (__bf16)win[i];
    if (i < KK * CPAD * DI) {
        const int k = i / (CPAD * DI);
        const int rem = i % (CPAD * DI);
        const int c = rem / DI, d = rem % DI;
        xpwbf[i] = (c < RK + 2 * NS) ? (__bf16)xpw[((size_t)k * 38 + c) * DI + d]
                                     : (__bf16)0.0f;
    }
    if (i < DM * DI) woutbf[i] = (__bf16)wout[i];
}

// ---------- kernel 2: in_proj GEMM  xz[8192,384] = Xbf[8192,96] @ Win[384,96]^T ----------
__global__ void gemm_inproj(const __bf16* __restrict__ A, const __bf16* __restrict__ Bt,
                            float* __restrict__ C) {
    const int tile = blockIdx.x * (blockDim.x >> 5) + (threadIdx.x >> 5);
    const int NT = (2 * DI) / 16;              // 24
    if (tile >= (MROWS / 16) * NT) return;     // wave-uniform
    const int nt = tile % NT, mt = tile / NT;
    v8f acc = wmma_tile_bf16<3>(A + (size_t)mt * 16 * DM, DM,
                                Bt + (size_t)nt * 16 * DM, DM);
    store_tile_f32(C + (size_t)(mt * 16) * (2 * DI) + nt * 16, 2 * DI, acc);
}

// ---------- kernel 3: depthwise 3x3 conv + bias + SiLU, scatter into 4 scan orders ----------
__global__ void conv_silu_build(const float* __restrict__ xz, const float* __restrict__ cw,
                                const float* __restrict__ cb, __bf16* __restrict__ xs) {
    const int i = blockIdx.x * blockDim.x + threadIdx.x;
    if (i >= BB * LL * DI) return;
    const int c = i % DI;
    const int pos = i / DI;
    const int w = pos % WW;
    const int h = (pos / WW) % HH;
    const int b = pos / (HH * WW);
    float s = cb[c];
#pragma unroll
    for (int kh = 0; kh < 3; ++kh) {
#pragma unroll
        for (int kw = 0; kw < 3; ++kw) {
            const int h2 = h + kh - 1, w2 = w + kw - 1;
            if (h2 >= 0 && h2 < HH && w2 >= 0 && w2 < WW)
                s += cw[c * 9 + kh * 3 + kw] *
                     xz[((size_t)b * LL + h2 * WW + w2) * (2 * DI) + c];
        }
    }
    const float v = s / (1.0f + __expf(-s));   // SiLU
    const __bf16 bv = (__bf16)v;
    const int l  = h * WW + w;
    const int lt = w * HH + h;
    const size_t base = (size_t)b * KK * LL * DI;
    xs[base + ((size_t)0 * LL + l) * DI + c]             = bv;
    xs[base + ((size_t)1 * LL + lt) * DI + c]            = bv;
    xs[base + ((size_t)2 * LL + (LL - 1 - l)) * DI + c]  = bv;
    xs[base + ((size_t)3 * LL + (LL - 1 - lt)) * DI + c] = bv;
}

// ---------- kernel 4: x_proj GEMM  xdbl[32768,48] = xs[32768,192] @ Wk[48,192]^T ----------
__global__ void gemm_xproj(const __bf16* __restrict__ xs, const __bf16* __restrict__ xpw,
                           float* __restrict__ xdbl) {
    const int tile = blockIdx.x * (blockDim.x >> 5) + (threadIdx.x >> 5);
    const int NT = CPAD / 16;                      // 3
    const int MT = (BB * KK * LL) / 16;            // 2048
    if (tile >= MT * NT) return;
    const int nt = tile % NT, mt = tile / NT;
    const int bk = (mt * 16) / LL;                 // rows of one tile share (b,k)
    const int k  = bk & 3;
    v8f acc = wmma_tile_bf16<6>(xs + (size_t)mt * 16 * DI, DI,
                                xpw + ((size_t)k * CPAD + nt * 16) * DI, DI);
    store_tile_f32(xdbl + (size_t)(mt * 16) * CPAD + nt * 16, CPAD, acc);
}

// ---------- kernel 5: selective scan, async-LDS double-buffered x_dbl rows ----------
// One wave per block; lane = state n (16 per half-wave), 2 channels per wave.
// Row l+1 (48 f32 = 12 lanes x b128) streams global->LDS on the ASYNC path while
// row l is consumed from LDS (same-address LDS reads broadcast).
__global__ void __launch_bounds__(32)
scan_kernel(const __bf16* __restrict__ xs, const float* __restrict__ xdbl,
            const float* __restrict__ dtw, const float* __restrict__ dtb,
            const float* __restrict__ A_logs, const float* __restrict__ Dsk,
            float* __restrict__ outy) {
    __shared__ __align__(16) float rowbuf[2][64];

    const int wave = blockIdx.x;               // 0 .. B*K*DI/2-1 = 767
    const int lane = threadIdx.x;
    const int n = lane & 15, half = lane >> 4;
    const int dpair = wave % (DI / 2);
    const int bk = wave / (DI / 2);
    const int k = bk & 3;
    const int d = dpair * 2 + half;

    const float An   = -__expf(A_logs[((size_t)k * DI + d) * NS + n]);
    const float bias = dtb[k * DI + d];
    const float Dk   = Dsk[k * DI + d];
    float wr[RK];
#pragma unroll
    for (int r = 0; r < RK; ++r) wr[r] = dtw[((size_t)k * DI + d) * RK + r];

    const float*  xd  = xdbl + (size_t)bk * LL * CPAD;
    const __bf16* u_p = xs   + (size_t)bk * LL * DI + d;
    float*        yo  = outy + (size_t)bk * LL * DI + d;

    // prime buffer 0 with row 0
#if HAVE_ASYNC_LDS
    if (lane < 12) async_copy16(xd + lane * 4, &rowbuf[0][lane * 4]);
#else
    if (lane < 12)
        *(float4*)&rowbuf[0][lane * 4] = *(const float4*)(xd + lane * 4);
#endif

    float h = 0.0f;
    for (int l = 0; l < LL; ++l) {
        const float* nrow = xd + (size_t)(l + 1) * CPAD;
#if HAVE_ASYNC_LDS
        if (l + 1 < LL) {
            if (lane < 12)
                async_copy16(nrow + lane * 4, &rowbuf[(l + 1) & 1][lane * 4]);
            wait_async<1>();   // async loads complete in order -> row l landed
        } else {
            wait_async<0>();
        }
#endif
        const float* cur = &rowbuf[l & 1][0];
        __builtin_prefetch(u_p + (size_t)(l + 1) * DI, 0, 1);

        float dtr = bias;
#pragma unroll
        for (int r = 0; r < RK; ++r) dtr += cur[r] * wr[r];   // LDS broadcast reads
        const float dt = (dtr > 20.0f) ? dtr : log1pf(__expf(dtr));   // softplus
        const float Bn = cur[RK + n];
        const float Cn = cur[RK + NS + n];
        const float u  = (float)u_p[(size_t)l * DI];
        const float dA = __expf(dt * An);
        h = h * dA + (dt * u) * Bn;

#if !HAVE_ASYNC_LDS
        if (l + 1 < LL && lane < 12)
            *(float4*)&rowbuf[(l + 1) & 1][lane * 4] = *(const float4*)(nrow + lane * 4);
#endif

        float p = h * Cn;
        p += __shfl_xor(p, 1, 32);
        p += __shfl_xor(p, 2, 32);
        p += __shfl_xor(p, 4, 32);
        p += __shfl_xor(p, 8, 32);             // sum over 16 states per half-wave
        if (n == 0) yo[(size_t)l * DI] = p + Dk * u;
    }
}

// ---------- kernel 6: combine 4 directions + LayerNorm + SiLU(z) gate -> bf16 ----------
__global__ void __launch_bounds__(256)
combine_ln_gate(const float* __restrict__ outy, const float* __restrict__ xz,
                const float* __restrict__ ln_g, const float* __restrict__ ln_b,
                __bf16* __restrict__ yln) {
    __shared__ float red[256];
    __shared__ float stats[2];
    const int bpos = blockIdx.x;               // 0..8191
    const int b = bpos / LL;
    const int l = bpos % LL;
    const int h = l / WW, w = l % WW;
    const int lt = w * HH + h;
    const int t = threadIdx.x;

    float val = 0.0f;
    if (t < DI) {
        const size_t base = (size_t)b * KK * LL * DI;
        val = outy[base + ((size_t)0 * LL + l) * DI + t]
            + outy[base + ((size_t)2 * LL + (LL - 1 - l)) * DI + t]
            + outy[base + ((size_t)1 * LL + lt) * DI + t]
            + outy[base + ((size_t)3 * LL + (LL - 1 - lt)) * DI + t];
    }
    red[t] = val; __syncthreads();
    for (int s = 128; s > 0; s >>= 1) { if (t < s) red[t] += red[t + s]; __syncthreads(); }
    if (t == 0) stats[0] = red[0] / (float)DI;
    __syncthreads();
    const float mu = stats[0];
    const float dv = (t < DI) ? (val - mu) : 0.0f;
    red[t] = dv * dv; __syncthreads();
    for (int s = 128; s > 0; s >>= 1) { if (t < s) red[t] += red[t + s]; __syncthreads(); }
    if (t == 0) stats[1] = red[0] / (float)DI;
    __syncthreads();
    if (t < DI) {
        const float yn = (val - mu) * rsqrtf(stats[1] + 1e-5f) * ln_g[t] + ln_b[t];
        const float z  = xz[((size_t)b * LL + l) * (2 * DI) + DI + t];
        const float g  = z / (1.0f + __expf(-z));     // SiLU
        yln[((size_t)b * LL + l) * DI + t] = (__bf16)(yn * g);
    }
}

// ---------- kernel 7: out_proj GEMM  out[8192,96] = yln[8192,192] @ Wout[96,192]^T ----------
__global__ void gemm_outproj(const __bf16* __restrict__ A, const __bf16* __restrict__ Bt,
                             float* __restrict__ C) {
    const int tile = blockIdx.x * (blockDim.x >> 5) + (threadIdx.x >> 5);
    const int NT = DM / 16;                    // 6
    if (tile >= (MROWS / 16) * NT) return;
    const int nt = tile % NT, mt = tile / NT;
    v8f acc = wmma_tile_bf16<6>(A + (size_t)mt * 16 * DI, DI,
                                Bt + (size_t)nt * 16 * DI, DI);
    store_tile_f32(C + (size_t)(mt * 16) * DM + nt * 16, DM, acc);
}

// ---------- launch ----------
extern "C" void kernel_launch(void* const* d_in, const int* in_sizes, int n_in,
                              void* d_out, int out_size, void* d_ws, size_t ws_size,
                              hipStream_t stream) {
    const float* x         = (const float*)d_in[0];
    const float* in_proj_w = (const float*)d_in[1];
    const float* conv_w    = (const float*)d_in[2];
    const float* conv_b    = (const float*)d_in[3];
    const float* x_proj_w  = (const float*)d_in[4];
    const float* dt_proj_w = (const float*)d_in[5];
    const float* dt_proj_b = (const float*)d_in[6];
    const float* A_logs    = (const float*)d_in[7];
    const float* Ds        = (const float*)d_in[8];
    const float* ln_g      = (const float*)d_in[9];
    const float* ln_b      = (const float*)d_in[10];
    const float* out_proj_w= (const float*)d_in[11];
    float* out = (float*)d_out;

    char* ws = (char*)d_ws;
    __bf16* xbf    = (__bf16*)(ws + OFF_XBF);
    __bf16* winbf  = (__bf16*)(ws + OFF_WIN);
    __bf16* xpwbf  = (__bf16*)(ws + OFF_XPW);
    __bf16* woutbf = (__bf16*)(ws + OFF_WOUT);
    float*  xz     = (float*) (ws + OFF_XZ);
    __bf16* xs     = (__bf16*)(ws + OFF_XS);
    float*  xdbl   = (float*) (ws + OFF_XDBL);
    float*  outy   = (float*) (ws + OFF_OUTY);
    __bf16* yln    = (__bf16*)(ws + OFF_YLN);

    // 1. cast to bf16 (+ pad x_proj_w 38->48)
    prep_cast<<<(MROWS * DM + 255) / 256, 256, 0, stream>>>(
        x, in_proj_w, x_proj_w, out_proj_w, xbf, winbf, xpwbf, woutbf);

    // 2. in_proj: 512 * 24 tiles, 8 waves/block -> 1536 blocks
    gemm_inproj<<<1536, 256, 0, stream>>>(xbf, winbf, xz);

    // 3. depthwise conv + SiLU + build 4 scan orderings
    conv_silu_build<<<(BB * LL * DI + 255) / 256, 256, 0, stream>>>(xz, conv_w, conv_b, xs);

    // 4. x_proj: 2048 * 3 tiles -> 768 blocks
    gemm_xproj<<<768, 256, 0, stream>>>(xs, xpwbf, xdbl);

    // 5. selective scan: 768 independent waves (B*K*DI/2), spread wide
    scan_kernel<<<BB * KK * (DI / 2), 32, 0, stream>>>(
        xs, xdbl, dt_proj_w, dt_proj_b, A_logs, Ds, outy);

    // 6. combine + layernorm + gate
    combine_ln_gate<<<BB * LL, 256, 0, stream>>>(outy, xz, ln_g, ln_b, yln);

    // 7. out_proj: 512 * 6 tiles -> 384 blocks
    gemm_outproj<<<384, 256, 0, stream>>>(yln, woutbf, out);
}
